// FrustumSegmentationNet_86990267613322
// MI455X (gfx1250) — matrix operand order repaired and checked by reading, not compiled, not run on hardware
//
#include <hip/hip_runtime.h>
#include <hip/hip_bf16.h>
#include <cstdint>

#define CROP   127
#define NCLS   82
#define HH     512
#define WW     512
#define BB     2
#define MM     4
#define NPTS   (CROP * CROP)       // 16129
#define NTILES 505                 // ceil(NPTS/32) row tiles of 32
#define NPAD   (NTILES * 32)       // 16160
#define NCHUNK 64
#define CHUNKN 253                 // ceil(NPTS/NCHUNK)

typedef _Float16 f16;
typedef __attribute__((ext_vector_type(16))) _Float16 v16h;
typedef __attribute__((ext_vector_type(8)))  _Float16 v8h;
typedef __attribute__((ext_vector_type(8)))  float    v8f;

union V16H { v16h v; v8h h[2]; f16 e[16]; };
union V8F  { v8f v; float e[8]; };

// ---------------------------------------------------------------- zero output
__global__ void zero_f32(float* __restrict__ p, long n) {
  long i = (long)blockIdx.x * blockDim.x + threadIdx.x;
  long stride = (long)gridDim.x * blockDim.x;
  for (; i < n; i += stride) p[i] = 0.0f;
}

// ------------------------------------------------- pack B into WMMA fragments
// Target layout (16x16x32 f16 B fragment, wave32):
//   lane l: col = nt*16 + (l&15); element e -> K = kt*32 + (e&7) + (l>=16 ? 8 : 0) + (e>=8 ? 16 : 0)
// Packed so each lane's v16h is contiguous (32 bytes).
__global__ void pack_b_f16(const float* __restrict__ W, f16* __restrict__ out,
                           int K, int Nc) {
  int g = blockIdx.x * blockDim.x + threadIdx.x;
  int total = K * Nc;
  if (g >= total) return;
  int KT   = K >> 5;
  int e    = g & 15;
  int lane = (g >> 4) & 31;
  int kt   = (g >> 9) % KT;
  int nt   = g / (KT << 9);
  int ke   = (e & 7) + ((lane & 16) >> 1) + ((e & 8) << 1);  // +8 for upper half-wave, +16 for e>=8
  int row  = (kt << 5) + ke;
  int col  = (nt << 4) + (lane & 15);
  out[g] = (f16)W[(size_t)row * Nc + col];
}

// ------------------------------------------------------------- 3x3 inverse
__device__ inline void inv3x3(const float* A, float* inv) {
  float a = A[0], b = A[1], c = A[2];
  float d = A[3], e = A[4], f = A[5];
  float g = A[6], h = A[7], i = A[8];
  float det = a * (e * i - f * h) - b * (d * i - f * g) + c * (d * h - e * g);
  float r = 1.0f / det;
  inv[0] = (e * i - f * h) * r; inv[1] = (c * h - b * i) * r; inv[2] = (b * f - c * e) * r;
  inv[3] = (f * g - d * i) * r; inv[4] = (a * i - c * g) * r; inv[5] = (c * d - a * f) * r;
  inv[6] = (d * h - e * g) * r; inv[7] = (b * g - a * h) * r; inv[8] = (a * e - b * d) * r;
}

// -------------------- build per-point features + layer1 (K=6) + ReLU -> f16
__global__ void build_hidden(const float* __restrict__ rgb,
                             const float* __restrict__ depth,
                             const float* __restrict__ intr,
                             const int* __restrict__ box,
                             const float* __restrict__ Wf1,
                             const float* __restrict__ bf1,
                             f16* __restrict__ hidden, int b, int m) {
  int n = blockIdx.x;          // one row per block, NPAD blocks
  int c = threadIdx.x;         // 256 output channels
  if (n >= NPTS) { hidden[(size_t)n * 256 + c] = (f16)0.0f; return; }
  int x1 = box[(b * MM + m) * 5 + 0];
  int y1 = box[(b * MM + m) * 5 + 1];
  int v = n / CROP, u = n % CROP;
  float Ki[9];
  inv3x3(intr + b * 9, Ki);
  float uu = u + 0.5f, vv = v + 0.5f;
  float px = Ki[0] * uu + Ki[1] * vv + Ki[2];
  float py = Ki[3] * uu + Ki[4] * vv + Ki[5];
  float pz = Ki[6] * uu + Ki[7] * vv + Ki[8];
  int gx = x1 + v, gy = y1 + u;
  float z = depth[((size_t)b * HH + gx) * WW + gy];
  const float* cp = rgb + (((size_t)b * HH + gx) * WW + gy) * 3;
  float x[6] = { px * z, py * z, pz * z, cp[0], cp[1], cp[2] };
  float s = bf1[c];
#pragma unroll
  for (int k = 0; k < 6; ++k) s += x[k] * Wf1[k * 256 + c];
  hidden[(size_t)n * 256 + c] = (f16)fmaxf(s, 0.0f);
}

// ----------------------------- WMMA GEMM: out = relu(A(NPADxK) @ W + bias)
// One wave per block; each wave computes a 32-row x 64-col output strip:
// 8 accumulators, 2 A fragments per K-step, each B fragment reused twice.
__global__ void __launch_bounds__(32)
gemm_wmma(const f16* __restrict__ A, const f16* __restrict__ Bp,
          const float* __restrict__ bias, f16* __restrict__ out,
          int K, int Nc) {
  int lane = threadIdx.x;
  int row0 = blockIdx.x << 5;
  int col0 = blockIdx.y << 6;
  int KT = K >> 5;

  V8F acc[8];
#pragma unroll
  for (int t = 0; t < 4; ++t) {
    float bv = bias[col0 + (t << 4) + (lane & 15)];
#pragma unroll
    for (int i = 0; i < 8; ++i) { acc[t].e[i] = bv; acc[4 + t].e[i] = bv; }
  }

  // A fragment base for this lane (16-bit A 16x32 layout: 8-half runs).
  const f16* arow0 = A + (size_t)(row0 + (lane & 15)) * K + ((lane >> 4) << 3);
  const f16* arow1 = arow0 + (size_t)16 * K;
  const f16* bbase = Bp + (size_t)(col0 >> 4) * KT * 512 + ((size_t)lane << 4);

  for (int kt = 0; kt < KT; ++kt) {
    V16H a0, a1;
    a0.h[0] = *(const v8h*)(arow0 + (kt << 5));
    a0.h[1] = *(const v8h*)(arow0 + (kt << 5) + 16);
    a1.h[0] = *(const v8h*)(arow1 + (kt << 5));
    a1.h[1] = *(const v8h*)(arow1 + (kt << 5) + 16);
    if (kt + 1 < KT) {
      __builtin_prefetch(arow0 + ((kt + 1) << 5), 0, 3);
      __builtin_prefetch(arow1 + ((kt + 1) << 5), 0, 3);
    }
#pragma unroll
    for (int t = 0; t < 4; ++t) {
      V16H bf;
      bf.v = *(const v16h*)(bbase + ((size_t)(t * KT + kt) << 9));
      acc[t].v = __builtin_amdgcn_wmma_f32_16x16x32_f16(
          false, a0.v, false, bf.v, (short)0, acc[t].v, false, false);
      acc[4 + t].v = __builtin_amdgcn_wmma_f32_16x16x32_f16(
          false, a1.v, false, bf.v, (short)0, acc[4 + t].v, false, false);
    }
  }

  // D layout: VGPR i, lane l -> row = rowbase + 8*(l>=16) + i, col = col0 + (l&15)
  int rbase = row0 + ((lane >> 4) << 3);
  int cb    = col0 + (lane & 15);
#pragma unroll
  for (int t = 0; t < 4; ++t)
#pragma unroll
    for (int i = 0; i < 8; ++i) {
      out[(size_t)(rbase + i) * Nc + cb + (t << 4)] =
          (f16)fmaxf(acc[t].e[i], 0.0f);
      out[(size_t)(rbase + 16 + i) * Nc + cb + (t << 4)] =
          (f16)fmaxf(acc[4 + t].e[i], 0.0f);
    }
}

// ------------------------------------------------ global max pool (chunked)
__global__ void maxpool_part(const f16* __restrict__ feat, float* __restrict__ gpart) {
  int col = blockIdx.x * 256 + threadIdx.x;   // grid.x = 4 -> 1024 cols
  int ch = blockIdx.y;
  int n0 = ch * CHUNKN;
  int n1 = n0 + CHUNKN; if (n1 > NPTS) n1 = NPTS;
  float mx = -3.0e38f;
  for (int n = n0; n < n1; ++n)
    mx = fmaxf(mx, (float)feat[(size_t)n * 1024 + col]);
  gpart[ch * 1024 + col] = mx;
}

__global__ void maxpool_reduce(const float* __restrict__ gpart, float* __restrict__ gfeat) {
  int col = blockIdx.x * 256 + threadIdx.x;
  float mx = -3.0e38f;
  for (int ch = 0; ch < NCHUNK; ++ch) mx = fmaxf(mx, gpart[ch * 1024 + col]);
  gfeat[col] = mx;
}

// ------------- lbl = gfeat@Wl + bl ; hconst = gfeat@Wh1[1024:] + bh1 (folded)
__global__ void head_vec(const float* __restrict__ gfeat,
                         const float* __restrict__ Wl, const float* __restrict__ bl,
                         const float* __restrict__ Wh1, const float* __restrict__ bh1,
                         float* __restrict__ lbl, float* __restrict__ hconst) {
  int t = threadIdx.x;  // 256 threads
  float s = bh1[t];
  for (int k = 0; k < 1024; ++k) s += gfeat[k] * Wh1[(size_t)(1024 + k) * 256 + t];
  hconst[t] = s;
  if (t < NCLS) {
    float s2 = bl[t];
    for (int k = 0; k < 1024; ++k) s2 += gfeat[k] * Wl[k * NCLS + t];
    lbl[t] = s2;
  }
}

// ------------- seg head (256->2), argmax mask, weighted scatter-add to label
__global__ void seg_scatter(const f16* __restrict__ h16,
                            const float* __restrict__ Wh2, const float* __restrict__ bh2,
                            const float* __restrict__ lbl, const int* __restrict__ box,
                            float* __restrict__ out, int b, int m) {
  int n = blockIdx.x * blockDim.x + threadIdx.x;
  if (n >= NPTS) return;
  float s0 = bh2[0], s1 = bh2[1];
  const f16* hr = h16 + (size_t)n * 256;
  for (int c = 0; c < 256; ++c) {
    float r = (float)hr[c];
    s0 += r * Wh2[c * 2 + 0];
    s1 += r * Wh2[c * 2 + 1];
  }
  if (s1 > s0) {   // argmax == 1 (ties pick class 0)
    int x1 = box[(b * MM + m) * 5 + 0];
    int y1 = box[(b * MM + m) * 5 + 1];
    int v = n / CROP, u = n % CROP;
    float dv = v - 63.5f, du = u - 63.5f;
    float kern = rsqrtf(dv * dv + du * du);
    float* dst = out + (size_t)b * NCLS * HH * WW + (size_t)(x1 + v) * WW + (y1 + u);
    for (int cls = 0; cls < NCLS; ++cls)
      atomicAdd(dst + (size_t)cls * HH * WW, lbl[cls] * kern);
  }
}

// ---------------------------------------------------------------- launcher
extern "C" void kernel_launch(void* const* d_in, const int* in_sizes, int n_in,
                              void* d_out, int out_size, void* d_ws, size_t ws_size,
                              hipStream_t stream) {
  const float* rgb   = (const float*)d_in[0];
  const float* depth = (const float*)d_in[1];
  const float* intr  = (const float*)d_in[2];
  const int*   box   = (const int*)d_in[3];
  const float* Wf1 = (const float*)d_in[4];
  const float* bf1 = (const float*)d_in[5];
  const float* Wf2 = (const float*)d_in[6];
  const float* bf2 = (const float*)d_in[7];
  const float* Wl  = (const float*)d_in[8];
  const float* bl  = (const float*)d_in[9];
  const float* Wh1 = (const float*)d_in[10];
  const float* bh1 = (const float*)d_in[11];
  const float* Wh2 = (const float*)d_in[12];
  const float* bh2 = (const float*)d_in[13];
  float* out = (float*)d_out;

  char* ws = (char*)d_ws;
  size_t o = 0;
  auto wsalloc = [&](size_t bytes) -> void* {
    void* p = ws + o;
    o = (o + bytes + 255) & ~(size_t)255;
    return p;
  };
  f16*   Bp2    = (f16*)wsalloc((size_t)256 * 1024 * sizeof(f16));
  f16*   Bp3    = (f16*)wsalloc((size_t)1024 * 256 * sizeof(f16));
  f16*   hidden = (f16*)wsalloc((size_t)NPAD * 256 * sizeof(f16));
  f16*   feat   = (f16*)wsalloc((size_t)NPAD * 1024 * sizeof(f16));
  f16*   h16    = (f16*)wsalloc((size_t)NPAD * 256 * sizeof(f16));
  float* gpart  = (float*)wsalloc((size_t)NCHUNK * 1024 * sizeof(float));
  float* gfeat  = (float*)wsalloc(1024 * sizeof(float));
  float* lblv   = (float*)wsalloc(128 * sizeof(float));
  float* hconst = (float*)wsalloc(256 * sizeof(float));
  if (o > ws_size) return;  // workspace too small (deterministic no-op)

  zero_f32<<<4096, 256, 0, stream>>>(out, (long)out_size);
  pack_b_f16<<<(256 * 1024 + 255) / 256, 256, 0, stream>>>(Wf2, Bp2, 256, 1024);
  pack_b_f16<<<(1024 * 256 + 255) / 256, 256, 0, stream>>>(Wh1, Bp3, 1024, 256);

  for (int b = 0; b < BB; ++b) {
    for (int m = 0; m < MM; ++m) {
      build_hidden<<<NPAD, 256, 0, stream>>>(rgb, depth, intr, box, Wf1, bf1,
                                             hidden, b, m);
      gemm_wmma<<<dim3(NTILES, 1024 / 64), 32, 0, stream>>>(hidden, Bp2, bf2,
                                                            feat, 256, 1024);
      maxpool_part<<<dim3(4, NCHUNK), 256, 0, stream>>>(feat, gpart);
      maxpool_reduce<<<4, 256, 0, stream>>>(gpart, gfeat);
      head_vec<<<1, 256, 0, stream>>>(gfeat, Wl, bl, Wh1, bh1, lblv, hconst);
      gemm_wmma<<<dim3(NTILES, 256 / 64), 32, 0, stream>>>(feat, Bp3, hconst,
                                                           h16, 1024, 256);
      seg_scatter<<<(NPTS + 127) / 128, 128, 0, stream>>>(h16, Wh2, bh2, lblv,
                                                          box, out, b, m);
    }
  }
}
